// GraphConvLayer_15564961481203
// MI455X (gfx1250) — matrix operand order repaired and checked by reading, not compiled, not run on hardware
//
#include <hip/hip_runtime.h>
#include <hip/hip_bf16.h>

#define NN   50000
#define NE   800000
#define D    256
#define EPSV 1e-6f

typedef __attribute__((ext_vector_type(16))) __bf16       v16bf;
typedef __attribute__((ext_vector_type(8)))  float        v8f;
typedef __attribute__((ext_vector_type(4)))  unsigned int u32x4;

union FragU { u32x4 q[2]; v16bf v; };

__device__ __forceinline__ unsigned short f2bf(float f) {
    unsigned int u = __builtin_bit_cast(unsigned int, f);
    return (unsigned short)((u + 0x7FFFu + ((u >> 16) & 1u)) >> 16);  // RNE
}

// ---------------- zero scratch (agg + counts contiguous) ----------------
__global__ __launch_bounds__(256) void zero_f32(float* __restrict__ p, int n) {
    int i = blockIdx.x * 256 + threadIdx.x;
    if (i < n) p[i] = 0.0f;
}

// ---- repack fp32 [K x 256] weights -> bf16 WMMA B-fragment order ----
// frag f = kt*16 + nt: k in [kt*32, kt*32+32), n in [nt*16, nt*16+16)
// lane l holds 16 bf16 at ushort offset f*512 + l*16 + j with
//   n = nt*16 + (l&15),  k = kt*32 + (l>>4)*8 + j + (j>=8 ? 8 : 0)      (ISA B 32x16 layout)
__global__ __launch_bounds__(256) void prep_weights(
        const float* __restrict__ Wm, const float* __restrict__ Wu,
        unsigned short* __restrict__ WbM, unsigned short* __restrict__ WbU) {
    int i = blockIdx.x * 256 + threadIdx.x;
    if (i < 768 * 256) {
        int f = i >> 9, r = i & 511, l = r >> 4, j = r & 15;
        int kt = f >> 4, nt = f & 15;
        int n = nt * 16 + (l & 15);
        int k = kt * 32 + (l >> 4) * 8 + j + ((j >> 3) << 3);
        WbM[i] = f2bf(Wm[k * 256 + n]);
    }
    if (i < 512 * 256) {
        int f = i >> 9, r = i & 511, l = r >> 4, j = r & 15;
        int kt = f >> 4, nt = f & 15;
        int n = nt * 16 + (l & 15);
        int k = kt * 32 + (l >> 4) * 8 + j + ((j >> 3) << 3);
        WbU[i] = f2bf(Wu[k * 256 + n]);
    }
}

// ---------------- edge message GEMM + scatter-aggregate ----------------
#define KMSG 768
#define SMSG 776   // padded LDS row stride (bf16 elems): 388 dwords == 4 mod 64 banks

__global__ __launch_bounds__(256) void msg_kernel(
        const float* __restrict__ nodef, const float* __restrict__ edgef,
        const int* __restrict__ eidx, const unsigned short* __restrict__ Wb,
        const float* __restrict__ bias,
        float* __restrict__ agg, float* __restrict__ cnt) {
    __shared__ unsigned short As[64 * SMSG];
    __shared__ int rowS[64];
    __shared__ int colS[64];

    const int tid = threadIdx.x;
    const int e0  = blockIdx.x * 64;

    if (tid < 64) {
        int r = eidx[e0 + tid];
        int c = eidx[NE + e0 + tid];
        rowS[tid] = r;
        colS[tid] = c;
        unsafeAtomicAdd(&cnt[c], 1.0f);   // exact integer-valued float sum
    }
    __syncthreads();

    // Gather + fp32->bf16 convert the A panel: [64 edges x 768] = [src|dst|ef]
    for (int i = tid; i < 64 * KMSG; i += 256) {
        int e = i / KMSG;
        int c = i - e * KMSG;
        float v;
        if (c < D)          v = nodef[(size_t)rowS[e] * D + c];
        else if (c < 2 * D) v = nodef[(size_t)colS[e] * D + (c - D)];
        else                v = edgef[(size_t)(e0 + e) * D + (c - 2 * D)];
        As[e * SMSG + c] = f2bf(v);
    }
    __syncthreads();

    const int lane  = tid & 31;
    const int wave  = tid >> 5;
    const int laneN = lane & 15;
    const int khalf = lane >> 4;

    v8f acc[4][2];
    #pragma unroll
    for (int mt = 0; mt < 4; ++mt) {
        acc[mt][0] = v8f{0.f,0.f,0.f,0.f,0.f,0.f,0.f,0.f};
        acc[mt][1] = v8f{0.f,0.f,0.f,0.f,0.f,0.f,0.f,0.f};
    }

    for (int kt = 0; kt < 24; ++kt) {
        FragU b0, b1;
        const u32x4* p0 = (const u32x4*)(Wb + (size_t)(kt * 16 + wave * 2 + 0) * 512);
        const u32x4* p1 = (const u32x4*)(Wb + (size_t)(kt * 16 + wave * 2 + 1) * 512);
        b0.q[0] = p0[lane * 2]; b0.q[1] = p0[lane * 2 + 1];
        b1.q[0] = p1[lane * 2]; b1.q[1] = p1[lane * 2 + 1];
        #pragma unroll
        for (int mt = 0; mt < 4; ++mt) {
            FragU a;   // ISA A 16x32 layout: K 0-7 & 16-23 (lanes 0-15) / 8-15 & 24-31
            const u32x4* pa = (const u32x4*)(As + (mt * 16 + laneN) * SMSG
                                                + kt * 32 + khalf * 8);
            a.q[0] = pa[0];
            a.q[1] = pa[2];   // +32 bytes
            acc[mt][0] = __builtin_amdgcn_wmma_f32_16x16x32_bf16(
                false, a.v, false, b0.v, (short)0, acc[mt][0], false, false);
            acc[mt][1] = __builtin_amdgcn_wmma_f32_16x16x32_bf16(
                false, a.v, false, b1.v, (short)0, acc[mt][1], false, false);
        }
    }

    // bias + ReLU + atomic scatter into agg[col]
    #pragma unroll
    for (int t = 0; t < 2; ++t) {
        int n = (wave * 2 + t) * 16 + laneN;
        float bv = bias[n];
        #pragma unroll
        for (int mt = 0; mt < 4; ++mt) {
            #pragma unroll
            for (int i = 0; i < 8; ++i) {
                int er = mt * 16 + khalf * 8 + i;   // C/D: M = vgpr + 8*lanehalf
                float v = acc[mt][t][i] + bv;
                v = v > 0.0f ? v : 0.0f;
                unsafeAtomicAdd(&agg[(size_t)colS[er] * D + n], v);
            }
        }
    }
}

// ---------------- node update GEMM ----------------
#define KUPD 512
#define SUPD 520   // 260 dwords == 4 mod 64 banks

__global__ __launch_bounds__(256) void upd_kernel(
        const float* __restrict__ nodef, const float* __restrict__ agg,
        const float* __restrict__ cnt, const unsigned short* __restrict__ Wb,
        const float* __restrict__ bias, float* __restrict__ out) {
    __shared__ unsigned short As[64 * SUPD];
    __shared__ float invS[64];

    const int tid = threadIdx.x;
    const int n0  = blockIdx.x * 64;

    if (tid < 64) {
        int node = n0 + tid; if (node >= NN) node = NN - 1;
        invS[tid] = 1.0f / (cnt[node] + EPSV);
    }
    __syncthreads();

    for (int i = tid; i < 64 * KUPD; i += 256) {
        int e = i >> 9;
        int c = i & 511;
        int node = n0 + e; if (node >= NN) node = NN - 1;   // clamp, keep EXEC full
        float v = (c < D) ? nodef[(size_t)node * D + c]
                          : agg[(size_t)node * D + (c - D)] * invS[e];
        As[e * SUPD + c] = f2bf(v);
    }
    __syncthreads();

    const int lane  = tid & 31;
    const int wave  = tid >> 5;
    const int laneN = lane & 15;
    const int khalf = lane >> 4;

    v8f acc[4][2];
    #pragma unroll
    for (int mt = 0; mt < 4; ++mt) {
        acc[mt][0] = v8f{0.f,0.f,0.f,0.f,0.f,0.f,0.f,0.f};
        acc[mt][1] = v8f{0.f,0.f,0.f,0.f,0.f,0.f,0.f,0.f};
    }

    for (int kt = 0; kt < 16; ++kt) {
        FragU b0, b1;
        const u32x4* p0 = (const u32x4*)(Wb + (size_t)(kt * 16 + wave * 2 + 0) * 512);
        const u32x4* p1 = (const u32x4*)(Wb + (size_t)(kt * 16 + wave * 2 + 1) * 512);
        b0.q[0] = p0[lane * 2]; b0.q[1] = p0[lane * 2 + 1];
        b1.q[0] = p1[lane * 2]; b1.q[1] = p1[lane * 2 + 1];
        #pragma unroll
        for (int mt = 0; mt < 4; ++mt) {
            FragU a;
            const u32x4* pa = (const u32x4*)(As + (mt * 16 + laneN) * SUPD
                                                + kt * 32 + khalf * 8);
            a.q[0] = pa[0];
            a.q[1] = pa[2];
            acc[mt][0] = __builtin_amdgcn_wmma_f32_16x16x32_bf16(
                false, a.v, false, b0.v, (short)0, acc[mt][0], false, false);
            acc[mt][1] = __builtin_amdgcn_wmma_f32_16x16x32_bf16(
                false, a.v, false, b1.v, (short)0, acc[mt][1], false, false);
        }
    }

    #pragma unroll
    for (int t = 0; t < 2; ++t) {
        int n = (wave * 2 + t) * 16 + laneN;
        float bv = bias[n];
        #pragma unroll
        for (int mt = 0; mt < 4; ++mt) {
            #pragma unroll
            for (int i = 0; i < 8; ++i) {
                int node = n0 + mt * 16 + khalf * 8 + i;
                float v = acc[mt][t][i] + bv;
                v = v > 0.0f ? v : 0.0f;
                if (node < NN) out[(size_t)node * D + n] = v;
            }
        }
    }
}

extern "C" void kernel_launch(void* const* d_in, const int* in_sizes, int n_in,
                              void* d_out, int out_size, void* d_ws, size_t ws_size,
                              hipStream_t stream) {
    const float* nodef = (const float*)d_in[0];
    const float* edgef = (const float*)d_in[1];
    const int*   eidx  = (const int*)d_in[2];
    const float* Wm    = (const float*)d_in[3];
    const float* bm    = (const float*)d_in[4];
    const float* Wu    = (const float*)d_in[5];
    const float* bu    = (const float*)d_in[6];
    float* out = (float*)d_out;

    // Workspace layout (bytes):
    //   [0, 393216)                  : W_msg bf16 fragments (24*16 frags * 1KB)
    //   [393216, 655360)             : W_upd bf16 fragments (16*16 frags * 1KB)
    //   [655360, 655360+NN*D*4)      : agg   (fp32, NN x D)
    //   [... , +NN*4)                : cnt   (fp32, NN)   -- contiguous with agg
    char* ws = (char*)d_ws;
    unsigned short* WbM = (unsigned short*)(ws);
    unsigned short* WbU = (unsigned short*)(ws + 393216);
    float* agg = (float*)(ws + 655360);
    float* cnt = agg + (size_t)NN * D;

    int nz = NN * D + NN;  // agg + cnt, contiguous
    zero_f32<<<(nz + 255) / 256, 256, 0, stream>>>(agg, nz);
    prep_weights<<<768, 256, 0, stream>>>(Wm, Wu, WbM, WbU);
    msg_kernel<<<NE / 64, 256, 0, stream>>>(nodef, edgef, eidx, WbM, bm, agg, cnt);
    upd_kernel<<<(NN + 63) / 64, 256, 0, stream>>>(nodef, agg, cnt, WbU, bu, out);
}
